// NNMFLayer_55765855371444
// MI455X (gfx1250) — compile-verified, matrix operand order checked by reading, not run on hardware
//
#include <hip/hip_runtime.h>

// ---------------------------------------------------------------------------
// NNMF layer for MI455X (gfx1250): bf16 WMMA GEMMs + f32 row-normalizations.
//   x: (B=2048, F_in=4096) f32     W: (F_out=2048, F_in=4096) f32
//   out: h (B, F_out) f32 after 10 multiplicative NNMF iterations.
// ---------------------------------------------------------------------------

typedef __attribute__((ext_vector_type(16))) __bf16 v16bf;
typedef __attribute__((ext_vector_type(8)))  __bf16 bf16x8;
typedef __attribute__((ext_vector_type(8)))  float  v8f;

#define SECURE_MIN 1e-5f
#define EPSF       1e-20f

#define BM 256
#define BN 128
#define BK 32
#define LDS_PAD 40   // bf16 elements per LDS row (40*2 = 80B, keeps 16B alignment)

// ---------------------------------------------------------------------------
// GEMM: C(MxN, f32, row-major) = A(MxK, bf16, row-major) * B(KxN)
// where B is supplied transposed: Bt(NxK, bf16, row-major).
// 256 threads = 8 waves; block tile 256x128; wave grid 4(M)x2(N);
// wave tile 64x64 = 4x4 wmma tiles -> 16 v_wmma per K-step, fed by
// 16 ds_load_b128 (1:1 LDS-load : WMMA issue ratio).
// ---------------------------------------------------------------------------
__global__ __launch_bounds__(256)
void gemm_bf16_nt(const __bf16* __restrict__ A,
                  const __bf16* __restrict__ Bt,
                  float* __restrict__ C,
                  int M, int N, int K)
{
    __shared__ __bf16 sA[BM * LDS_PAD];   // 20480 B
    __shared__ __bf16 sB[BN * LDS_PAD];   // 10240 B

    const int tid  = threadIdx.x;
    const int lane = tid & 31;
    const int wid  = tid >> 5;      // 0..7
    const int wm   = wid >> 1;      // 0..3 : 64-row slab
    const int wn   = wid & 1;       // 0..1 : 64-col slab

    const int block_m = blockIdx.y * BM;
    const int block_n = blockIdx.x * BN;

    // Global->LDS copy mapping:
    //   A tile 256x32: thread t moves row t, 32 bf16 (4 x b128).
    //   B tile 128x32: thread t moves row t>>1, 16 bf16 (2 x b128).
    const int brow = tid >> 1;
    const int bseg = (tid & 1) * 16;

    const __bf16* Arow = A  + (size_t)(block_m + tid)  * (size_t)K;
    const __bf16* Brow = Bt + (size_t)(block_n + brow) * (size_t)K + bseg;
    __bf16* sArow = &sA[tid * LDS_PAD];
    __bf16* sBrow = &sB[brow * LDS_PAD + bseg];

    // WMMA fragment lane mapping (16-bit A 16x32 layout, ISA 7.12.2):
    // lanes 0-15 hold K=0..7 (vgpr0-3) and K=16..23 (vgpr4-7); lanes 16-31
    // hold K=8..15 and K=24..31. half*8 selects the per-lane K origin.
    const int half = lane >> 4;           // 0 or 1
    const int mr   = lane & 15;           // row (A) / col (B) within 16-tile

    union Frag { v16bf v; bf16x8 h[2]; };

    v8f acc[4][4];
    const v8f vzero = {0.f, 0.f, 0.f, 0.f, 0.f, 0.f, 0.f, 0.f};
    #pragma unroll
    for (int i = 0; i < 4; ++i)
        #pragma unroll
        for (int j = 0; j < 4; ++j)
            acc[i][j] = vzero;

    for (int k0 = 0; k0 < K; k0 += BK) {
        bf16x8 a0 = *(const bf16x8*)(Arow + k0);
        bf16x8 a1 = *(const bf16x8*)(Arow + k0 + 8);
        bf16x8 a2 = *(const bf16x8*)(Arow + k0 + 16);
        bf16x8 a3 = *(const bf16x8*)(Arow + k0 + 24);
        bf16x8 b0 = *(const bf16x8*)(Brow + k0);
        bf16x8 b1 = *(const bf16x8*)(Brow + k0 + 8);
        if (k0 + BK < K) {                       // hint next tiles into cache
            __builtin_prefetch(Arow + k0 + BK, 0, 3);
            __builtin_prefetch(Brow + k0 + BK, 0, 3);
        }
        __syncthreads();                         // LDS reuse from previous step
        *(bf16x8*)(sArow)      = a0;
        *(bf16x8*)(sArow + 8)  = a1;
        *(bf16x8*)(sArow + 16) = a2;
        *(bf16x8*)(sArow + 24) = a3;
        *(bf16x8*)(sBrow)      = b0;
        *(bf16x8*)(sBrow + 8)  = b1;
        __syncthreads();

        Frag afr[4], bfr[4];
        #pragma unroll
        for (int i = 0; i < 4; ++i) {
            const int row = (wm * 64 + i * 16 + mr) * LDS_PAD;
            afr[i].h[0] = *(const bf16x8*)(&sA[row + half * 8]);
            afr[i].h[1] = *(const bf16x8*)(&sA[row + 16 + half * 8]);
        }
        #pragma unroll
        for (int j = 0; j < 4; ++j) {
            const int col = (wn * 64 + j * 16 + mr) * LDS_PAD;
            bfr[j].h[0] = *(const bf16x8*)(&sB[col + half * 8]);
            bfr[j].h[1] = *(const bf16x8*)(&sB[col + 16 + half * 8]);
        }

        #pragma unroll
        for (int i = 0; i < 4; ++i)
            #pragma unroll
            for (int j = 0; j < 4; ++j)
                acc[i][j] = __builtin_amdgcn_wmma_f32_16x16x32_bf16(
                    false, afr[i].v, false, bfr[j].v,
                    (short)0, acc[i][j], false, false);
    }

    // C/D f32 16x16 layout: vgpr r holds M = r (lanes 0-15) or M = r+8 (16-31),
    // N = lane & 15.
    const int row_off = (lane >> 4) * 8;
    const int ccol    = lane & 15;
    #pragma unroll
    for (int i = 0; i < 4; ++i) {
        #pragma unroll
        for (int j = 0; j < 4; ++j) {
            const int mbase = block_m + wm * 64 + i * 16 + row_off;
            const int nbase = block_n + wn * 64 + j * 16 + ccol;
            float* Cp = C + (size_t)mbase * (size_t)N + nbase;
            #pragma unroll
            for (int r = 0; r < 8; ++r)
                Cp[(size_t)r * (size_t)N] = acc[i][j][r];
        }
    }
}

// ---------------------------------------------------------------------------
// Row-wise helpers. One block (256 threads) per row.
// ---------------------------------------------------------------------------
__device__ __forceinline__ float block_reduce_sum(float v, float* red)
{
    red[threadIdx.x] = v;
    __syncthreads();
    for (int off = 128; off > 0; off >>= 1) {
        if ((int)threadIdx.x < off) red[threadIdx.x] += red[threadIdx.x + off];
        __syncthreads();
    }
    return red[0];
}

// clip(W,SECURE_MIN), L1-normalize each row; emit bf16 in both layouts.
__global__ __launch_bounds__(256)
void prep_w(const float* __restrict__ W,
            __bf16* __restrict__ Wn,    // F_out x F_in
            __bf16* __restrict__ WnT,   // F_in x F_out
            int F_out, int F_in)
{
    __shared__ float red[256];
    const int row = blockIdx.x;
    const float* w = W + (size_t)row * F_in;
    float s = 0.f;
    for (int c = threadIdx.x; c < F_in; c += 256) s += fmaxf(w[c], SECURE_MIN);
    const float sum = block_reduce_sum(s, red);
    const float inv = 1.0f / fmaxf(sum, EPSF);
    for (int c = threadIdx.x; c < F_in; c += 256) {
        const __bf16 b = (__bf16)(fmaxf(w[c], SECURE_MIN) * inv);
        Wn [(size_t)row * F_in  + c]   = b;
        WnT[(size_t)c   * F_out + row] = b;
    }
}

// L1-normalize rows of x -> xn (f32, kept full precision for the ratio).
__global__ __launch_bounds__(256)
void prep_x(const float* __restrict__ x, float* __restrict__ xn, int F_in)
{
    __shared__ float red[256];
    const int row = blockIdx.x;
    const float* xr = x + (size_t)row * F_in;
    float s = 0.f;
    for (int c = threadIdx.x; c < F_in; c += 256) s += fabsf(xr[c]);
    const float sum = block_reduce_sum(s, red);
    const float inv = 1.0f / fmaxf(sum, EPSF);
    for (int c = threadIdx.x; c < F_in; c += 256)
        xn[(size_t)row * F_in + c] = xr[c] * inv;
}

__global__ __launch_bounds__(256)
void init_h(float* __restrict__ hf, __bf16* __restrict__ hb, int total, float val)
{
    const int i = blockIdx.x * blockDim.x + threadIdx.x;
    if (i < total) { hf[i] = val; hb[i] = (__bf16)val; }
}

// ratio = xn / (clip(recon)/max(rowsum,eps)) = xn * rowsum / clip(recon) -> bf16
__global__ __launch_bounds__(256)
void ratio_kernel(const float* __restrict__ recon,
                  const float* __restrict__ xn,
                  __bf16* __restrict__ ratio, int F_in)
{
    __shared__ float red[256];
    const int row = blockIdx.x;
    const float* rr = recon + (size_t)row * F_in;
    const float* xr = xn    + (size_t)row * F_in;
    float s = 0.f;
    for (int c = threadIdx.x; c < F_in; c += 256) s += fmaxf(rr[c], SECURE_MIN);
    const float scale = fmaxf(block_reduce_sum(s, red), EPSF);
    for (int c = threadIdx.x; c < F_in; c += 256)
        ratio[(size_t)row * F_in + c] =
            (__bf16)(xr[c] * scale / fmaxf(rr[c], SECURE_MIN));
}

// h = L1rownorm(clip(h*update)); writes f32 (h_out) and bf16 mirror.
__global__ __launch_bounds__(256)
void update_h(const float* __restrict__ upd,
              const float* __restrict__ h_in,
              float* __restrict__ h_out,
              __bf16* __restrict__ h_bf, int F_out)
{
    __shared__ float red[256];
    const int row = blockIdx.x;
    const size_t base = (size_t)row * F_out;
    float s = 0.f;
    for (int c = threadIdx.x; c < F_out; c += 256)
        s += fmaxf(h_in[base + c] * upd[base + c], SECURE_MIN);
    const float inv = 1.0f / fmaxf(block_reduce_sum(s, red), EPSF);
    for (int c = threadIdx.x; c < F_out; c += 256) {
        const float v = fmaxf(h_in[base + c] * upd[base + c], SECURE_MIN) * inv;
        h_out[base + c] = v;
        h_bf [base + c] = (__bf16)v;
    }
}

// ---------------------------------------------------------------------------
extern "C" void kernel_launch(void* const* d_in, const int* in_sizes, int n_in,
                              void* d_out, int out_size, void* d_ws, size_t ws_size,
                              hipStream_t stream)
{
    (void)in_sizes; (void)n_in; (void)out_size; (void)ws_size;
    const int B = 2048, F_in = 4096, F_out = 2048;
    const int N_ITER = 10;

    const float* x = (const float*)d_in[0];   // (B, F_in)
    const float* W = (const float*)d_in[1];   // (F_out, F_in)

    // Workspace carve-out (256B aligned). Total ~136 MB.
    char*  ws  = (char*)d_ws;
    size_t off = 0;
    auto carve = [&](size_t bytes) {
        char* p = ws + off;
        off += (bytes + 255) & ~(size_t)255;
        return (void*)p;
    };
    __bf16* Wn    = (__bf16*)carve((size_t)F_out * F_in * 2);  // 16 MB
    __bf16* WnT   = (__bf16*)carve((size_t)F_in * F_out * 2);  // 16 MB
    float*  xn    = (float*) carve((size_t)B * F_in * 4);      // 32 MB
    float*  h_f32 = (float*) carve((size_t)B * F_out * 4);     // 16 MB
    __bf16* h_bf  = (__bf16*)carve((size_t)B * F_out * 2);     //  8 MB
    float*  recon = (float*) carve((size_t)B * F_in * 4);      // 32 MB
    __bf16* ratio = (__bf16*)carve((size_t)B * F_in * 2);      // 16 MB
    float*  upd   = recon;  // alias: recon is dead once ratio is built

    prep_w<<<F_out, 256, 0, stream>>>(W, Wn, WnT, F_out, F_in);
    prep_x<<<B, 256, 0, stream>>>(x, xn, F_in);
    init_h<<<(B * F_out + 255) / 256, 256, 0, stream>>>(
        h_f32, h_bf, B * F_out, 1.0f / (float)F_out);

    const dim3 grid1(F_in  / BN, B / BM);   // recon  : M=B,  N=F_in,  K=F_out
    const dim3 grid2(F_out / BN, B / BM);   // update : M=B,  N=F_out, K=F_in

    for (int it = 0; it < N_ITER; ++it) {
        // recon = h @ Wn   (Bt = WnT: F_in rows x F_out cols)
        gemm_bf16_nt<<<grid1, 256, 0, stream>>>(h_bf, WnT, recon, B, F_in, F_out);
        ratio_kernel<<<B, 256, 0, stream>>>(recon, xn, ratio, F_in);
        // update = ratio @ Wn^T   (Bt = Wn: F_out rows x F_in cols)
        gemm_bf16_nt<<<grid2, 256, 0, stream>>>(ratio, Wn, upd, B, F_out, F_in);
        float* hout = (it == N_ITER - 1) ? (float*)d_out : h_f32;
        update_h<<<B, 256, 0, stream>>>(upd, h_f32, hout, h_bf, F_out);
    }
}
